// EncoderLayer_54692113547725
// MI455X (gfx1250) — matrix operand order repaired and checked by reading, not compiled, not run on hardware
//
#include <hip/hip_runtime.h>

// ---------------------------------------------------------------------------
// Problem constants (from reference)
// ---------------------------------------------------------------------------
#define D_MODEL 128
#define NHEAD   8
#define DHEAD   16
#define DFF     512
#define W1n     4000
#define L1n     64
#define V1n     48
#define W2n     500
#define L2n     256
#define V2n     216
#define NTOK    300000   // W1*V1 + W2*V2

// Weight-fragment workspace layout (in halves). Each fragment = 32 lanes * 16 halves = 512.
#define OFF_QKV 0                 // in_proj: 24 col-tiles * 4 k-chunks = 96 frags
#define OFF_OUT (96 * 512)        // out_proj: 8 * 4 = 32 frags
#define OFF_L1  (128 * 512)       // lin1: 32 * 4 = 128 frags
#define OFF_L2  (256 * 512)       // lin2: 8 * 16 = 128 frags
#define NFRAGS  384

typedef __attribute__((ext_vector_type(16))) _Float16 v16h;
typedef __attribute__((ext_vector_type(8)))  _Float16 v8h;
typedef __attribute__((ext_vector_type(8)))  float    v8f;

#define DEVINL __device__ __forceinline__

// ---------------------------------------------------------------------------
// WMMA helpers (CDNA5 f16 16x16x32, f32 accumulate)
// ---------------------------------------------------------------------------
DEVINL v8f wmma16(v16h a, v16h b, v8f c) {
  return __builtin_amdgcn_wmma_f32_16x16x32_f16(false, a, false, b, (short)0, c,
                                                false, false);
}

// A fragment: 16x32 f16 tile, row-major in LDS with stride ldk (halves).
// Contiguous k order used here: lanes 0-15 -> M=lane, K in kb..kb+15, kb=0;
//             lanes 16-31 -> M=lane-16, kb=16.
DEVINL v16h load_a16(const _Float16* a, int ldk) {
  int l  = threadIdx.x & 31;
  int m  = l & 15;
  int kb = (l < 16) ? 0 : 16;
  const _Float16* p = a + m * ldk + kb;
  v8h lo = *(const v8h*)(p);       // k = kb .. kb+7
  v8h hi = *(const v8h*)(p + 8);   // k = kb+8 .. kb+15
  return __builtin_shufflevector(lo, hi, 0, 1, 2, 3, 4, 5, 6, 7, 8, 9, 10, 11,
                                 12, 13, 14, 15);
}

// A fragment where only K=0..15 is real (K 16..31 forced to zero). ldk in halves.
DEVINL v16h load_a16_k16(const _Float16* a, int ldk) {
  int l  = threadIdx.x & 31;
  int m  = l & 15;
  v8h lo = *(const v8h*)(a + m * ldk);
  v8h hi = *(const v8h*)(a + m * ldk + 8);
  if (l >= 16) { lo = (v8h){}; hi = (v8h){}; }
  return __builtin_shufflevector(lo, hi, 0, 1, 2, 3, 4, 5, 6, 7, 8, 9, 10, 11,
                                 12, 13, 14, 15);
}

// B fragment: 32x16 f16. lanes 0-15 -> N=lane, K=0..15; lanes 16-31 -> K=16..31.
// Source here is row-major [key][16] with only 16 real K rows (upper lanes -> 0).
DEVINL v16h load_b16_k16(const _Float16* kh) {
  int l = threadIdx.x & 31;
  const _Float16* p = kh + (l & 15) * 16;
  v8h lo = *(const v8h*)(p);
  v8h hi = *(const v8h*)(p + 8);
  _Float16 f = (l < 16) ? (_Float16)1.0f : (_Float16)0.0f;
  lo *= f;
  hi *= f;
  return __builtin_shufflevector(lo, hi, 0, 1, 2, 3, 4, 5, 6, 7, 8, 9, 10, 11,
                                 12, 13, 14, 15);
}

// B fragment from column-major storage: vT[n * ldn + k], full K=0..31 chunk.
DEVINL v16h load_b16_colmaj(const _Float16* vT, int ldn) {
  int l   = threadIdx.x & 31;
  int n   = l & 15;
  int kl  = (l < 16) ? 0 : 16;
  const _Float16* p = vT + n * ldn + kl;
  v8h lo = *(const v8h*)(p);
  v8h hi = *(const v8h*)(p + 8);
  return __builtin_shufflevector(lo, hi, 0, 1, 2, 3, 4, 5, 6, 7, 8, 9, 10, 11,
                                 12, 13, 14, 15);
}

// Packed weight fragment from workspace (exact per-lane image, 32B per lane).
DEVINL v16h load_b16_ws(const _Float16* frag) {
  return *(const v16h*)(frag + (threadIdx.x & 31) * 16);
}

// ---------------------------------------------------------------------------
// Kernel 0: repack f32 weights into per-lane f16 B-fragment images in ws.
// One wave per fragment. 384 fragments total.
// ---------------------------------------------------------------------------
__global__ void pack_frags(const float* __restrict__ inw,
                           const float* __restrict__ outw,
                           const float* __restrict__ l1w,
                           const float* __restrict__ l2w,
                           _Float16* __restrict__ frags) {
  int gw   = (int)((blockIdx.x * blockDim.x + threadIdx.x) >> 5);
  int lane = threadIdx.x & 31;
  if (gw >= NFRAGS) return;
  const float* W;
  int in_f, nkc, baseo, fid;
  if (gw < 96) {
    W = inw;  in_f = 128; nkc = 4;  baseo = OFF_QKV; fid = gw;
  } else if (gw < 128) {
    W = outw; in_f = 128; nkc = 4;  baseo = OFF_OUT; fid = gw - 96;
  } else if (gw < 256) {
    W = l1w;  in_f = 128; nkc = 4;  baseo = OFF_L1;  fid = gw - 128;
  } else {
    W = l2w;  in_f = 512; nkc = 16; baseo = OFF_L2;  fid = gw - 256;
  }
  int ct = fid / nkc, kc = fid % nkc;
  int n  = lane & 15;
  int kl = (lane < 16) ? 0 : 16;
  const float* p = W + (size_t)(ct * 16 + n) * in_f + kc * 32 + kl;
  _Float16* q = frags + baseo + fid * 512 + lane * 16;
#pragma unroll
  for (int hh = 0; hh < 16; ++hh) q[hh] = (_Float16)p[hh];
}

// ---------------------------------------------------------------------------
// Kernel 1: fused per-window attention (gather + QKV + softmax attn + out-proj
// + scatter). One block per window, one wave per 16-query tile.
// ---------------------------------------------------------------------------
template <int L>
__global__ void attn_window(const float* __restrict__ src,
                            const float* __restrict__ pos,
                            const int*   __restrict__ idx,
                            const float* __restrict__ inb,
                            const float* __restrict__ outb,
                            const _Float16* __restrict__ frags,
                            float* __restrict__ out, int V) {
  constexpr int NW = L / 16;   // waves per block
  constexpr int NT = NW * 32;  // threads per block
  constexpr int KT = L / 16;   // key tiles

  extern __shared__ float4 smem4[];
  _Float16* qkh  = (_Float16*)smem4;        // [L][128] feat+pos (q,k input)
  _Float16* xh   = qkh + L * 128;           // [L][128] feat (v input)
  _Float16* obuf = xh + L * 128;            // [L][128] attn output (pre out-proj)
  _Float16* Kh   = obuf + L * 128;          // [L][16]  per-head K, row-major
  _Float16* VhT  = Kh + L * 16;             // [16][L]  per-head V, transposed
  _Float16* pscr = VhT + L * 16;            // [NW][16][32] P staging
  _Float16* qscr = pscr + NW * 512;         // [NW][16][16] Q staging

  const int w    = blockIdx.x;
  const int tid  = threadIdx.x;
  const int lane = tid & 31;
  const int wv   = tid >> 5;            // wave id == query-tile id
  const int n    = lane & 15;           // C-matrix column
  const int hl   = (lane < 16) ? 0 : 8; // C-matrix row offset

  // ---- gather: x = src[idx], qk = x + pos ----
  const float4* src4 = (const float4*)src;
  const float4* pos4 = (const float4*)(pos + (size_t)w * L * 128);
  for (int e = tid; e < L * 32; e += NT) {
    int t = e >> 5, c = e & 31;
    int tok  = idx[w * L + t];
    float4 s = src4[(size_t)tok * 32 + c];
    float4 p = pos4[t * 32 + c];
    int o = t * 128 + c * 4;
    xh[o + 0]  = (_Float16)s.x;        xh[o + 1]  = (_Float16)s.y;
    xh[o + 2]  = (_Float16)s.z;        xh[o + 3]  = (_Float16)s.w;
    qkh[o + 0] = (_Float16)(s.x + p.x); qkh[o + 1] = (_Float16)(s.y + p.y);
    qkh[o + 2] = (_Float16)(s.z + p.z); qkh[o + 3] = (_Float16)(s.w + p.w);
  }
  __syncthreads();

  const _Float16* fq = frags + OFF_QKV;

  for (int h = 0; h < NHEAD; ++h) {
    // ---- per-head Q/K/V tiles (each wave makes 16 rows) ----
    v8f cq = {}, ck = {}, cv = {};
#pragma unroll
    for (int kc = 0; kc < 4; ++kc) {
      v16h aqk = load_a16(qkh + wv * 16 * 128 + kc * 32, 128);
      v16h ax  = load_a16(xh + wv * 16 * 128 + kc * 32, 128);
      v16h bq  = load_b16_ws(fq + ((h)*4 + kc) * 512);       // Q rows 0..127
      v16h bk  = load_b16_ws(fq + ((8 + h) * 4 + kc) * 512); // K rows 128..255
      v16h bv  = load_b16_ws(fq + ((16 + h) * 4 + kc) * 512);// V rows 256..383
      cq = wmma16(aqk, bq, cq);
      ck = wmma16(aqk, bk, ck);
      cv = wmma16(ax, bv, cv);
    }
    float bqs = inb[h * 16 + n];
    float bks = inb[128 + h * 16 + n];
    float bvs = inb[256 + h * 16 + n];
#pragma unroll
    for (int r = 0; r < 8; ++r) {
      int t = wv * 16 + r + hl;
      Kh[t * 16 + n]               = (_Float16)(ck[r] + bks);
      VhT[n * L + t]               = (_Float16)(cv[r] + bvs);
      qscr[wv * 256 + (r + hl) * 16 + n] = (_Float16)(cq[r] + bqs);
    }
    __syncthreads();

    // ---- scores S = Q K^T / 4 with key padding mask ----
    v16h aq = load_a16_k16(qscr + wv * 256, 16);
    float sc[KT][8];
#pragma unroll
    for (int kt = 0; kt < KT; ++kt) {
      v16h bk = load_b16_k16(Kh + kt * 16 * 16);
      v8f z = {};
      v8f s = wmma16(aq, bk, z);
      int key = kt * 16 + n;
#pragma unroll
      for (int r = 0; r < 8; ++r)
        sc[kt][r] = (key < V) ? s[r] * 0.25f : -1e30f;
    }
    // ---- softmax over keys (rows live per C-reg within 16-lane halves) ----
#pragma unroll
    for (int r = 0; r < 8; ++r) {
      float mx = -1e30f;
#pragma unroll
      for (int kt = 0; kt < KT; ++kt) mx = fmaxf(mx, sc[kt][r]);
#pragma unroll
      for (int d = 1; d < 16; d <<= 1) mx = fmaxf(mx, __shfl_xor(mx, d, 32));
      float sum = 0.f;
#pragma unroll
      for (int kt = 0; kt < KT; ++kt) {
        float e = __expf(sc[kt][r] - mx);
        sc[kt][r] = e;
        sum += e;
      }
#pragma unroll
      for (int d = 1; d < 16; d <<= 1) sum += __shfl_xor(sum, d, 32);
      float inv = 1.f / sum;
#pragma unroll
      for (int kt = 0; kt < KT; ++kt) sc[kt][r] *= inv;
    }
    // ---- O = P V (stage P tiles through per-wave LDS for A-layout) ----
    v8f oacc = {};
#pragma unroll
    for (int kc2 = 0; kc2 < L / 32; ++kc2) {
#pragma unroll
      for (int half = 0; half < 2; ++half) {
        int kt = kc2 * 2 + half;
#pragma unroll
        for (int r = 0; r < 8; ++r)
          pscr[wv * 512 + (r + hl) * 32 + half * 16 + n] = (_Float16)sc[kt][r];
      }
      v16h ap = load_a16(pscr + wv * 512, 32);
      v16h bv = load_b16_colmaj(VhT + kc2 * 32, L);
      oacc = wmma16(ap, bv, oacc);
    }
#pragma unroll
    for (int r = 0; r < 8; ++r)
      obuf[(wv * 16 + r + hl) * 128 + h * 16 + n] = (_Float16)oacc[r];
    __syncthreads();
  }

  // ---- out projection + scatter (valid tokens only; indices are disjoint) ----
  const _Float16* fo = frags + OFF_OUT;
#pragma unroll
  for (int ct = 0; ct < 8; ++ct) {
    v8f c = {};
#pragma unroll
    for (int kc = 0; kc < 4; ++kc) {
      v16h a = load_a16(obuf + wv * 16 * 128 + kc * 32, 128);
      v16h b = load_b16_ws(fo + (ct * 4 + kc) * 512);
      c = wmma16(a, b, c);
    }
    float bo = outb[ct * 16 + n];
#pragma unroll
    for (int r = 0; r < 8; ++r) {
      int tl = wv * 16 + r + hl;
      if (tl < V) {
        int tok = idx[w * L + tl];
        out[(size_t)tok * 128 + ct * 16 + n] = c[r] + bo;
      }
    }
  }
}

// ---------------------------------------------------------------------------
// Kernel 2: x = LayerNorm(src + src2), in place in io (d_out).
// 16 tokens per block; 16 lanes per token (shuffle-reduce within half-waves).
// ---------------------------------------------------------------------------
__global__ void ln1_kernel(const float* __restrict__ src,
                           const float* __restrict__ w,
                           const float* __restrict__ b,
                           float* __restrict__ io) {
  int t   = blockIdx.x * 16 + (threadIdx.x >> 4);
  int l16 = threadIdx.x & 15;
  float v[8];
  float s = 0.f;
#pragma unroll
  for (int i = 0; i < 8; ++i) {
    int d = l16 + 16 * i;
    v[i]  = src[(size_t)t * 128 + d] + io[(size_t)t * 128 + d];
    s += v[i];
  }
#pragma unroll
  for (int d2 = 1; d2 < 16; d2 <<= 1) s += __shfl_xor(s, d2, 32);
  float mean = s * (1.f / 128.f);
  float q = 0.f;
#pragma unroll
  for (int i = 0; i < 8; ++i) {
    float dd = v[i] - mean;
    q += dd * dd;
  }
#pragma unroll
  for (int d2 = 1; d2 < 16; d2 <<= 1) q += __shfl_xor(q, d2, 32);
  float rstd = rsqrtf(q * (1.f / 128.f) + 1e-5f);
#pragma unroll
  for (int i = 0; i < 8; ++i) {
    int d = l16 + 16 * i;
    io[(size_t)t * 128 + d] = (v[i] - mean) * rstd * w[d] + b[d];
  }
}

// ---------------------------------------------------------------------------
// Kernel 3: fused FFN + residual + LN2, in place in io (d_out).
// 16 tokens per block, 8 waves; hidden tile kept in LDS (f16).
// ---------------------------------------------------------------------------
__global__ void ffn_kernel(const float* __restrict__ l1b,
                           const float* __restrict__ l2b,
                           const float* __restrict__ ln2w,
                           const float* __restrict__ ln2b,
                           const _Float16* __restrict__ frags,
                           float* __restrict__ io) {
  __shared__ _Float16 xh[16 * 128];
  __shared__ _Float16 hid[16 * 512];
  __shared__ float    xf[16 * 128];
  __shared__ float    zb[16 * 128];

  int tid = threadIdx.x, lane = tid & 31, wv = tid >> 5;
  int n = lane & 15, hl = (lane < 16) ? 0 : 8;
  size_t base = (size_t)blockIdx.x * 16 * 128;

  for (int e = tid; e < 2048; e += 256) {
    float x = io[base + e];
    xf[e] = x;
    xh[e] = (_Float16)x;
  }
  __syncthreads();

  // lin1 + ReLU: [16,128] x [128,512]
  const _Float16* f1 = frags + OFF_L1;
#pragma unroll
  for (int j = 0; j < 4; ++j) {
    int ct = wv + 8 * j;
    v8f c = {};
#pragma unroll
    for (int kc = 0; kc < 4; ++kc) {
      v16h a = load_a16(xh + kc * 32, 128);
      v16h b = load_b16_ws(f1 + (ct * 4 + kc) * 512);
      c = wmma16(a, b, c);
    }
    float bb = l1b[ct * 16 + n];
#pragma unroll
    for (int r = 0; r < 8; ++r)
      hid[(r + hl) * 512 + ct * 16 + n] = (_Float16)fmaxf(c[r] + bb, 0.f);
  }
  __syncthreads();

  // lin2 + residual: [16,512] x [512,128]
  const _Float16* f2 = frags + OFF_L2;
  {
    int ct = wv;
    v8f c = {};
#pragma unroll
    for (int kc = 0; kc < 16; ++kc) {
      v16h a = load_a16(hid + kc * 32, 512);
      v16h b = load_b16_ws(f2 + (ct * 16 + kc) * 512);
      c = wmma16(a, b, c);
    }
    float bb = l2b[ct * 16 + n];
#pragma unroll
    for (int r = 0; r < 8; ++r) {
      int m = r + hl;
      zb[m * 128 + ct * 16 + n] = xf[m * 128 + ct * 16 + n] + c[r] + bb;
    }
  }
  __syncthreads();

  // LN2
  int t16 = tid >> 4, l16 = tid & 15;
  float v[8];
  float s = 0.f;
#pragma unroll
  for (int i = 0; i < 8; ++i) {
    v[i] = zb[t16 * 128 + l16 + 16 * i];
    s += v[i];
  }
#pragma unroll
  for (int d2 = 1; d2 < 16; d2 <<= 1) s += __shfl_xor(s, d2, 32);
  float mean = s * (1.f / 128.f);
  float q = 0.f;
#pragma unroll
  for (int i = 0; i < 8; ++i) {
    float dd = v[i] - mean;
    q += dd * dd;
  }
#pragma unroll
  for (int d2 = 1; d2 < 16; d2 <<= 1) q += __shfl_xor(q, d2, 32);
  float rstd = rsqrtf(q * (1.f / 128.f) + 1e-5f);
#pragma unroll
  for (int i = 0; i < 8; ++i) {
    int d = l16 + 16 * i;
    io[base + t16 * 128 + d] = (v[i] - mean) * rstd * ln2w[d] + ln2b[d];
  }
}

// ---------------------------------------------------------------------------
// Launcher
// ---------------------------------------------------------------------------
static inline size_t attn_smem_bytes(int L) {
  return (size_t)(3 * L * 128 + 2 * L * 16 + (L / 16) * 768) * 2;
}

extern "C" void kernel_launch(void* const* d_in, const int* in_sizes, int n_in,
                              void* d_out, int out_size, void* d_ws,
                              size_t ws_size, hipStream_t stream) {
  (void)in_sizes; (void)n_in; (void)out_size; (void)ws_size;
  const float* src   = (const float*)d_in[0];
  const float* pos1  = (const float*)d_in[1];
  const float* pos2  = (const float*)d_in[2];
  const float* in_w  = (const float*)d_in[3];
  const float* in_b  = (const float*)d_in[4];
  const float* out_w = (const float*)d_in[5];
  const float* out_b = (const float*)d_in[6];
  const float* l1w   = (const float*)d_in[7];
  const float* l1b   = (const float*)d_in[8];
  const float* l2w   = (const float*)d_in[9];
  const float* l2b   = (const float*)d_in[10];
  const float* ln1w  = (const float*)d_in[11];
  const float* ln1b  = (const float*)d_in[12];
  const float* ln2w  = (const float*)d_in[13];
  const float* ln2b  = (const float*)d_in[14];
  const int*   idx1  = (const int*)d_in[15];
  const int*   idx2  = (const int*)d_in[16];
  float* out = (float*)d_out;
  _Float16* frags = (_Float16*)d_ws;

  // 1) pack weights into WMMA B-fragment images (384 waves).
  pack_frags<<<48, 256, 0, stream>>>(in_w, out_w, l1w, l2w, frags);

  // 2) windowed attention, both groups; writes src2 directly into d_out.
  attn_window<L1n><<<W1n, (L1n / 16) * 32, attn_smem_bytes(L1n), stream>>>(
      src, pos1, idx1, in_b, out_b, frags, out, V1n);
  attn_window<L2n><<<W2n, (L2n / 16) * 32, attn_smem_bytes(L2n), stream>>>(
      src, pos2, idx2, in_b, out_b, frags, out, V2n);

  // 3) x = LN1(src + src2), in place.
  ln1_kernel<<<NTOK / 16, 256, 0, stream>>>(src, ln1w, ln1b, out);

  // 4) fused FFN + residual + LN2, in place.
  ffn_kernel<<<NTOK / 16, 256, 0, stream>>>(l1b, l2b, ln2w, ln2b, frags, out);
}
